// AttentionLayer_52201032515806
// MI455X (gfx1250) — matrix operand order repaired and checked by reading, not compiled, not run on hardware
//
#include <hip/hip_runtime.h>
#include <hip/hip_bf16.h>
#include <math.h>

typedef _Float16 f16;
typedef _Float16 v16h __attribute__((ext_vector_type(16)));
typedef _Float16 v8h  __attribute__((ext_vector_type(8)));
typedef float    v8f  __attribute__((ext_vector_type(8)));

// async-to-LDS builtin expects int4 pointers: AS1 (global) src, AS3 (LDS) dst
typedef int v4i __attribute__((vector_size(16)));
typedef __attribute__((address_space(1))) v4i g_v4i;
typedef __attribute__((address_space(3))) v4i l_v4i;

#if __has_builtin(__builtin_amdgcn_global_load_async_to_lds_b128)
#define HAVE_ASYNC_LDS 1
#else
#define HAVE_ASYNC_LDS 0
#endif

static __device__ __forceinline__ void wait_async0() {
#if __has_builtin(__builtin_amdgcn_s_wait_asynccnt)
  __builtin_amdgcn_s_wait_asynccnt(0);
#else
  asm volatile("s_wait_asynccnt 0" ::: "memory");
#endif
}
static __device__ __forceinline__ void wait_async2() {
#if __has_builtin(__builtin_amdgcn_s_wait_asynccnt)
  __builtin_amdgcn_s_wait_asynccnt(2);
#else
  asm volatile("s_wait_asynccnt 2" ::: "memory");
#endif
}

#define WMMA_F16(A, B, C) \
  __builtin_amdgcn_wmma_f32_16x16x32_f16(false, (A), false, (B), (short)0, (C), false, false)

// ---- fragment loaders (CDNA5 wave32 WMMA 16x16x32 f16 layouts) ----
// A 16x32: lane<16 -> row M=lane, K {0..7,16..23}; lane>=16 -> K {8..15,24..31}
static __device__ __forceinline__ v16h ldA(const f16* row, int k0, int hi) {
  union { v16h v; v8h h[2]; } u;
  u.h[0] = *(const v8h*)(row + k0 + hi * 8);
  u.h[1] = *(const v8h*)(row + k0 + 16 + hi * 8);
  return u.v;
}
// B 32x16: lane<16 -> col N=lane, K {0..7,16..23}; lane>=16 -> col N=lane-16, K {8..15,24..31}
static __device__ __forceinline__ v16h ldB(const f16* col, int k0, int hi) {
  union { v16h v; v8h h[2]; } u;
  u.h[0] = *(const v8h*)(col + k0 + hi * 8);
  u.h[1] = *(const v8h*)(col + k0 + 16 + hi * 8);
  return u.v;
}

// ---------------- f32 -> f16 conversion ----------------
__global__ void k_f32_to_f16(const float* __restrict__ src, f16* __restrict__ dst, int n) {
  int i = blockIdx.x * blockDim.x + threadIdx.x;
  if (i < n) dst[i] = (f16)src[i];
}

// ---------------- QKV GEMM: [8192x512] x [512x1536], 32x64 tile per wave ----------------
__global__ __launch_bounds__(256) void k_qkv_gemm(const f16* __restrict__ xh,
                                                  const f16* __restrict__ wh,
                                                  f16* __restrict__ qh,
                                                  f16* __restrict__ kh,
                                                  f16* __restrict__ vh) {
  const int NT = 24;  // 1536/64 column tiles
  int wid  = blockIdx.x * 8 + (threadIdx.x >> 5);   // 768 blocks * 8 = 6144 waves
  int lane = threadIdx.x & 31, lo = lane & 15, hi = lane >> 4;
  int mt = wid / NT, nt = wid % NT;
  int mb = mt * 32, nb = nt * 64;
  const f16* ar0 = xh + (size_t)(mb + lo) * 512;
  const f16* ar1 = xh + (size_t)(mb + 16 + lo) * 512;
  const f16* bc0 = wh + (size_t)(nb + 0  + lo) * 512;  // B col n = qkv_w row n
  const f16* bc1 = wh + (size_t)(nb + 16 + lo) * 512;
  const f16* bc2 = wh + (size_t)(nb + 32 + lo) * 512;
  const f16* bc3 = wh + (size_t)(nb + 48 + lo) * 512;
  v8f c0[4] = {}, c1[4] = {};
  for (int k = 0; k < 512; k += 32) {
    v16h a0 = ldA(ar0, k, hi);
    v16h a1 = ldA(ar1, k, hi);
    v16h b0 = ldB(bc0, k, hi);
    v16h b1 = ldB(bc1, k, hi);
    v16h b2 = ldB(bc2, k, hi);
    v16h b3 = ldB(bc3, k, hi);
    c0[0] = WMMA_F16(a0, b0, c0[0]);
    c1[0] = WMMA_F16(a1, b0, c1[0]);
    c0[1] = WMMA_F16(a0, b1, c0[1]);
    c1[1] = WMMA_F16(a1, b1, c1[1]);
    c0[2] = WMMA_F16(a0, b2, c0[2]);
    c1[2] = WMMA_F16(a1, b2, c1[2]);
    c0[3] = WMMA_F16(a0, b3, c0[3]);
    c1[3] = WMMA_F16(a1, b3, c1[3]);
  }
  // 64-wide column block lies inside one (q/k/v, head) region: d spans 0..63
  int which = nb >> 9;
  int h     = (nb & 511) >> 6;
  f16* dst = (which == 0) ? qh : (which == 1 ? kh : vh);
#pragma unroll
  for (int half = 0; half < 2; ++half) {
#pragma unroll
    for (int r = 0; r < 8; ++r) {
      int mg = mb + half * 16 + r + 8 * hi;   // global row = b*N + n
      int bb = mg >> 11, nn = mg & 2047;
      size_t rb = (((size_t)(bb * 8 + h)) * 2048 + nn) * 64 + lo;
      const v8f* cc = half ? c1 : c0;
#pragma unroll
      for (int j = 0; j < 4; ++j) dst[rb + j * 16] = (f16)cc[j][r];
    }
  }
}

// ---------------- in-place NeoX RoPE on q,k: pair (d, d+32) ----------------
__global__ void k_rope(f16* __restrict__ qh, f16* __restrict__ kh) {
  int tid = blockIdx.x * blockDim.x + threadIdx.x;  // B*H*N*32 threads
  int j  = tid & 31;
  int n  = (tid >> 5) & 2047;
  int bh = tid >> 16;
  size_t base = ((size_t)bh * 2048 + n) * 64;
  float inv = __expf(-(float)j * (9.210340371976184f / 32.0f));  // 10000^(-j/32)
  float f = (float)n * inv;
  float s, cc;
  __sincosf(f, &s, &cc);
  float q1 = (float)qh[base + j], q2 = (float)qh[base + j + 32];
  qh[base + j]      = (f16)(q1 * cc - q2 * s);
  qh[base + j + 32] = (f16)(q2 * cc + q1 * s);
  float k1 = (float)kh[base + j], k2 = (float)kh[base + j + 32];
  kh[base + j]      = (f16)(k1 * cc - k2 * s);
  kh[base + j + 32] = (f16)(k2 * cc + k1 * s);
}

// ---------------- flash attention: block = (b,h, 128 queries) = 8 waves x 16 queries ----------------
// Async path: double-buffered cache->LDS DMA (ASYNCcnt), DMA of tile i+1 overlaps compute of tile i.
__global__ __launch_bounds__(256) void k_flash(const f16* __restrict__ qh,
                                               const f16* __restrict__ kh,
                                               const f16* __restrict__ vh,
                                               f16* __restrict__ oh) {
#if HAVE_ASYNC_LDS
  __shared__ alignas(16) f16 sK[2][32 * 64];   // K tiles, row-major [key][d], double-buffered
  __shared__ alignas(16) f16 sVr[2][32 * 64];  // V tiles row-major (async dest), double-buffered
#else
  __shared__ alignas(16) f16 sK1[32 * 64];
#endif
  __shared__ alignas(16) f16 sVt[64 * 32];     // V tile transposed [d][key]
  __shared__ alignas(16) f16 sP[8][16 * 32];   // per-wave P staging (D-layout -> A-layout)

  int bh = blockIdx.x >> 4;   // b*8+h, 0..31
  int qb = blockIdx.x & 15;
  int h  = bh & 7;
  int wid = threadIdx.x >> 5, lane = threadIdx.x & 31;
  int lo = lane & 15, hi = lane >> 4;
  int qt = qb * 128 + wid * 16;

  const f16* qbase = qh + ((size_t)bh * 2048 + qt) * 64;
  const f16* krow0 = kh + (size_t)bh * 2048 * 64;
  const f16* vrow0 = vh + (size_t)bh * 2048 * 64;

  // Q A-fragments (d contraction 64 = 2 chunks of 32)
  v16h aq0 = ldA(qbase + (size_t)lo * 64, 0, hi);
  v16h aq1 = ldA(qbase + (size_t)lo * 64, 32, hi);

  v8f o0 = {}, o1 = {}, o2 = {}, o3 = {};
  float mrow[8], lrow[8];
#pragma unroll
  for (int r = 0; r < 8; ++r) { mrow[r] = -1e30f; lrow[r] = 0.0f; }

  const float scale = 0.125f;                    // D^-0.5
  const float slope = exp2f((float)(2 - h));     // alibi slope[h] * MAX_BIAS = 2^(2-h)

  int ck = threadIdx.x >> 3;        // cooperative loader: key-local 0..31
  int cd = (threadIdx.x & 7) * 8;   // d offset 0..56

#if HAVE_ASYNC_LDS
  // prologue: issue DMA for tile 0
  __builtin_amdgcn_global_load_async_to_lds_b128(
      (g_v4i*)(krow0 + (size_t)ck * 64 + cd), (l_v4i*)(&sK[0][0] + ck * 64 + cd), 0, 0);
  __builtin_amdgcn_global_load_async_to_lds_b128(
      (g_v4i*)(vrow0 + (size_t)ck * 64 + cd), (l_v4i*)(&sVr[0][0] + ck * 64 + cd), 0, 0);
#endif

  for (int kb = 0; kb < 2048; kb += 32) {
#if HAVE_ASYNC_LDS
    int cur = (kb >> 5) & 1;
    __syncthreads();   // all waves done reading buffers about to be overwritten
    if (kb + 32 < 2048) {
      int nxt = cur ^ 1;
      __builtin_amdgcn_global_load_async_to_lds_b128(
          (g_v4i*)(krow0 + (size_t)(kb + 32 + ck) * 64 + cd),
          (l_v4i*)(&sK[nxt][0] + ck * 64 + cd), 0, 0);
      __builtin_amdgcn_global_load_async_to_lds_b128(
          (g_v4i*)(vrow0 + (size_t)(kb + 32 + ck) * 64 + cd),
          (l_v4i*)(&sVr[nxt][0] + ck * 64 + cd), 0, 0);
      wait_async2();   // async loads complete in order: tile `cur` (2 older ops) has landed
    } else {
      wait_async0();
    }
    __syncthreads();   // every wave's pieces of tile `cur` visible
    // in-LDS transpose of V for PV B-fragments
    {
      v8h vv = *(const v8h*)(&sVr[cur][0] + ck * 64 + cd);
#pragma unroll
      for (int i = 0; i < 8; ++i) sVt[(cd + i) * 32 + ck] = vv[i];
    }
    __syncthreads();
    const f16* sKc = &sK[cur][0];
#else
    __syncthreads();
    v8h kv = *(const v8h*)(krow0 + (size_t)(kb + ck) * 64 + cd);
    *(v8h*)(sK1 + ck * 64 + cd) = kv;
    v8h vv = *(const v8h*)(vrow0 + (size_t)(kb + ck) * 64 + cd);
#pragma unroll
    for (int i = 0; i < 8; ++i) sVt[(cd + i) * 32 + ck] = vv[i];
    if (kb + 32 < 2048) {
      __builtin_prefetch(krow0 + (size_t)(kb + 32 + ck) * 64 + cd, 0, 0);
      __builtin_prefetch(vrow0 + (size_t)(kb + 32 + ck) * 64 + cd, 0, 0);
    }
    __syncthreads();
    const f16* sKc = sK1;
#endif

    // S = Q K^T for 32 keys (two 16-key sub-tiles)
    v8f s0 = {}, s1 = {};
    {
      v16h b00 = ldB(sKc + lo * 64, 0, hi);
      v16h b01 = ldB(sKc + lo * 64, 32, hi);
      s0 = WMMA_F16(aq0, b00, s0);
      s0 = WMMA_F16(aq1, b01, s0);
      v16h b10 = ldB(sKc + (16 + lo) * 64, 0, hi);
      v16h b11 = ldB(sKc + (16 + lo) * 64, 32, hi);
      s1 = WMMA_F16(aq0, b10, s1);
      s1 = WMMA_F16(aq1, b11, s1);
    }

    // scale + ALiBi + online softmax (rows spread over vgpr idx, lane-half)
    float p0[8], p1[8], alpha[8];
    int key0 = kb + lo, key1 = kb + 16 + lo;
#pragma unroll
    for (int r = 0; r < 8; ++r) {
      int qg = qt + r + 8 * hi;
      float r0 = (float)(key0 - qg); if (r0 > 0.0f) r0 = 0.0f;
      float r1 = (float)(key1 - qg); if (r1 > 0.0f) r1 = 0.0f;
      float v0 = s0[r] * scale + slope * r0;
      float v1 = s1[r] * scale + slope * r1;
      float mx = fmaxf(v0, v1);
      mx = fmaxf(mx, __shfl_xor(mx, 1, 32));
      mx = fmaxf(mx, __shfl_xor(mx, 2, 32));
      mx = fmaxf(mx, __shfl_xor(mx, 4, 32));
      mx = fmaxf(mx, __shfl_xor(mx, 8, 32));
      float mnew = fmaxf(mrow[r], mx);
      alpha[r] = __expf(mrow[r] - mnew);
      p0[r] = __expf(v0 - mnew);
      p1[r] = __expf(v1 - mnew);
      float ls = p0[r] + p1[r];
      ls += __shfl_xor(ls, 1, 32);
      ls += __shfl_xor(ls, 2, 32);
      ls += __shfl_xor(ls, 4, 32);
      ls += __shfl_xor(ls, 8, 32);
      lrow[r] = lrow[r] * alpha[r] + ls;
      mrow[r] = mnew;
    }
#pragma unroll
    for (int r = 0; r < 8; ++r) {
      o0[r] *= alpha[r]; o1[r] *= alpha[r]; o2[r] *= alpha[r]; o3[r] *= alpha[r];
    }

    // P: D-layout -> LDS -> A-layout (wave-private region; LDS ops in-order per wave)
    f16* pw = &sP[wid][0];
#pragma unroll
    for (int r = 0; r < 8; ++r) {
      pw[(r + 8 * hi) * 32 + lo]      = (f16)p0[r];
      pw[(r + 8 * hi) * 32 + 16 + lo] = (f16)p1[r];
    }
    v16h pa = ldA(pw + lo * 32, 0, hi);

    // O += P @ V  (K=32 keys in one WMMA; 4 tiles cover d=0..63)
    v16h bv0 = ldB(sVt + (0  + lo) * 32, 0, hi);
    v16h bv1 = ldB(sVt + (16 + lo) * 32, 0, hi);
    v16h bv2 = ldB(sVt + (32 + lo) * 32, 0, hi);
    v16h bv3 = ldB(sVt + (48 + lo) * 32, 0, hi);
    o0 = WMMA_F16(pa, bv0, o0);
    o1 = WMMA_F16(pa, bv1, o1);
    o2 = WMMA_F16(pa, bv2, o2);
    o3 = WMMA_F16(pa, bv3, o3);
  }

  // epilogue: normalize, write [B,N,C] f16
  int b = bh >> 3;
#pragma unroll
  for (int r = 0; r < 8; ++r) {
    float inv = 1.0f / lrow[r];
    int qg = qt + r + 8 * hi;
    size_t rowoff = ((size_t)(b * 2048 + qg)) * 512 + h * 64;
    oh[rowoff + 0  + lo] = (f16)(o0[r] * inv);
    oh[rowoff + 16 + lo] = (f16)(o1[r] * inv);
    oh[rowoff + 32 + lo] = (f16)(o2[r] * inv);
    oh[rowoff + 48 + lo] = (f16)(o3[r] * inv);
  }
}

// ---------------- projection GEMM: [8192x512] x [512x512] + bias, 32x64 tile per wave ----------------
__global__ __launch_bounds__(256) void k_proj_gemm(const f16* __restrict__ oh,
                                                   const f16* __restrict__ pwh,
                                                   const float* __restrict__ pb,
                                                   float* __restrict__ out) {
  const int NT = 8;  // 512/64
  int wid  = blockIdx.x * 8 + (threadIdx.x >> 5);   // 256 blocks * 8 = 2048 waves
  int lane = threadIdx.x & 31, lo = lane & 15, hi = lane >> 4;
  int mt = wid / NT, nt = wid % NT;
  int mb = mt * 32, nb = nt * 64;
  const f16* ar0 = oh + (size_t)(mb + lo) * 512;
  const f16* ar1 = oh + (size_t)(mb + 16 + lo) * 512;
  const f16* bc0 = pwh + (size_t)(nb + 0  + lo) * 512;
  const f16* bc1 = pwh + (size_t)(nb + 16 + lo) * 512;
  const f16* bc2 = pwh + (size_t)(nb + 32 + lo) * 512;
  const f16* bc3 = pwh + (size_t)(nb + 48 + lo) * 512;
  v8f c0[4] = {}, c1[4] = {};
  for (int k = 0; k < 512; k += 32) {
    v16h a0 = ldA(ar0, k, hi);
    v16h a1 = ldA(ar1, k, hi);
    v16h b0 = ldB(bc0, k, hi);
    v16h b1 = ldB(bc1, k, hi);
    v16h b2 = ldB(bc2, k, hi);
    v16h b3 = ldB(bc3, k, hi);
    c0[0] = WMMA_F16(a0, b0, c0[0]);
    c1[0] = WMMA_F16(a1, b0, c1[0]);
    c0[1] = WMMA_F16(a0, b1, c0[1]);
    c1[1] = WMMA_F16(a1, b1, c1[1]);
    c0[2] = WMMA_F16(a0, b2, c0[2]);
    c1[2] = WMMA_F16(a1, b2, c1[2]);
    c0[3] = WMMA_F16(a0, b3, c0[3]);
    c1[3] = WMMA_F16(a1, b3, c1[3]);
  }
#pragma unroll
  for (int half = 0; half < 2; ++half) {
#pragma unroll
    for (int r = 0; r < 8; ++r) {
      int mg = mb + half * 16 + r + 8 * hi;
      const v8f* cc = half ? c1 : c0;
#pragma unroll
      for (int j = 0; j < 4; ++j) {
        out[(size_t)mg * 512 + nb + j * 16 + lo] = cc[j][r] + pb[nb + j * 16 + lo];
      }
    }
  }
}

extern "C" void kernel_launch(void* const* d_in, const int* in_sizes, int n_in,
                              void* d_out, int out_size, void* d_ws, size_t ws_size,
                              hipStream_t stream) {
  const float* x      = (const float*)d_in[0];  // [4,2048,512]
  const float* qkv_w  = (const float*)d_in[1];  // [1536,512]
  const float* proj_w = (const float*)d_in[2];  // [512,512]
  const float* proj_b = (const float*)d_in[3];  // [512]

  // workspace layout (f16 elements)
  f16* ws = (f16*)d_ws;
  size_t off = 0;
  f16* xh  = ws + off; off += (size_t)8192 * 512;        // x in f16
  f16* wh  = ws + off; off += (size_t)1536 * 512;        // qkv_w in f16
  f16* pwh = ws + off; off += (size_t)512 * 512;         // proj_w in f16
  f16* qh  = ws + off; off += (size_t)32 * 2048 * 64;    // q [B*H,N,D]
  f16* kh  = ws + off; off += (size_t)32 * 2048 * 64;    // k
  f16* vh  = ws + off; off += (size_t)32 * 2048 * 64;    // v
  f16* oh  = ws + off; off += (size_t)8192 * 512;        // attn out [B,N,C]

  k_f32_to_f16<<<(8192 * 512) / 256, 256, 0, stream>>>(x, xh, 8192 * 512);
  k_f32_to_f16<<<(1536 * 512) / 256, 256, 0, stream>>>(qkv_w, wh, 1536 * 512);
  k_f32_to_f16<<<(512 * 512) / 256, 256, 0, stream>>>(proj_w, pwh, 512 * 512);

  // 256 M-tiles x 24 N-tiles = 6144 waves / 8 per block
  k_qkv_gemm<<<768, 256, 0, stream>>>(xh, wh, qh, kh, vh);

  // B*H*N*32 = 2097152 threads
  k_rope<<<8192, 256, 0, stream>>>(qh, kh);

  // 4*8*16 = 512 blocks (b,h, 128-query slabs)
  k_flash<<<512, 256, 0, stream>>>(qh, kh, vh, oh);

  // 256 M-tiles x 8 N-tiles = 2048 waves / 8 per block
  k_proj_gemm<<<256, 256, 0, stream>>>(oh, pwh, proj_b, (float*)d_out);
}
